// GoldenTransformer_128849019286
// MI455X (gfx1250) — compile-verified, matrix operand order checked by reading, not run on hardware
//
#include <hip/hip_runtime.h>

// ---------------------------------------------------------------- types ----
typedef __attribute__((ext_vector_type(16))) __bf16 v16bf;
typedef __attribute__((ext_vector_type(8)))  __bf16 v8bf;
typedef __attribute__((ext_vector_type(8)))  float  v8f;

// ------------------------------------------------------------ elementwise --
__global__ void k_f32_to_bf16(const float* __restrict__ s, __bf16* __restrict__ d, int n) {
    int t = blockIdx.x * blockDim.x + threadIdx.x;
    if (t < n) d[t] = (__bf16)s[t];
}

__global__ void k_copy_f32(const float* __restrict__ s, float* __restrict__ d, int n) {
    int t = blockIdx.x * blockDim.x + threadIdx.x;
    if (t < n) d[t] = s[t];
}

__global__ void k_fill_u32(unsigned* __restrict__ p, unsigned v, int n) {
    int t = blockIdx.x * blockDim.x + threadIdx.x;
    if (t < n) p[t] = v;
}

// Pack W [K x Nw] (row-major f32) -> Wt [Nw x K] (row-major bf16).
__global__ void k_pack_wt(const float* __restrict__ W, __bf16* __restrict__ Wt, int K, int Nw) {
    int t = blockIdx.x * blockDim.x + threadIdx.x;
    if (t >= K * Nw) return;
    int c = t / K, k = t - c * K;
    Wt[(size_t)c * K + k] = (__bf16)W[(size_t)k * Nw + c];
}

// ------------------------------------------------------------- WMMA GEMM ---
// D[M x Nout] = A[M x K] (bf16, row-major) * Bt[Nout x K]^T (bf16) + bias.
// Each wave computes a 16x64 output strip: one A fragment is reused across
// 4 accumulators. All 4 B fragments are staged into distinct registers BEFORE
// any WMMA issues, so the loads form one clause and the 4 WMMAs run
// back-to-back against the next iteration's load latency.
// Block = 128 threads (4 waves) -> 16 rows x 256 cols.
// grid.x = M/16, grid.y = ceil(Nout/256).
template <bool OUT_BF16>
__global__ void k_wmma_gemm(const __bf16* __restrict__ A, const __bf16* __restrict__ Bt,
                            const float* __restrict__ bias,
                            float* __restrict__ Df, __bf16* __restrict__ Db,
                            int M, int K, int Nout, int ldD) {
    const int lane   = threadIdx.x & 31;
    const int wave   = threadIdx.x >> 5;
    const int tileM  = blockIdx.x;
    const int tileN0 = blockIdx.y * 16 + wave * 4;     // first of 4 col tiles
    const int nTiles = Nout >> 4;
    if (tileM * 16 >= M || tileN0 >= nTiles) return;

    const int half = lane >> 4;     // fragment K half
    const int rc   = lane & 15;     // A-row (M) / B-column (N) / D-column

    const __bf16* arow = A + (size_t)(tileM * 16 + rc) * K;
    const __bf16* brow[4];
#pragma unroll
    for (int j = 0; j < 4; ++j) {
        int tN = tileN0 + j; if (tN >= nTiles) tN = nTiles - 1;   // clamp (loads only)
        brow[j] = Bt + (size_t)(tN * 16 + rc) * K;
    }

    v8f acc[4] = {{}, {}, {}, {}};
    for (int k0 = 0; k0 < K; k0 += 32) {
        const int ko = k0 + half * 8;
        // ---- stage A and all 4 B fragments (distinct registers, one clause) ----
        v8bf alo = *(const v8bf*)(arow + ko);
        v8bf ahi = *(const v8bf*)(arow + ko + 16);
        v8bf blo[4], bhi[4];
#pragma unroll
        for (int j = 0; j < 4; ++j) {
            blo[j] = *(const v8bf*)(brow[j] + ko);
            bhi[j] = *(const v8bf*)(brow[j] + ko + 16);
        }
        if (k0 + 32 < K) __builtin_prefetch((const void*)(arow + k0 + 32), 0, 3);
        v16bf a;
#pragma unroll
        for (int i = 0; i < 8; ++i) { a[i] = alo[i]; a[i + 8] = ahi[i]; }
        // ---- 4 independent WMMAs ----
#pragma unroll
        for (int j = 0; j < 4; ++j) {
            v16bf b;
#pragma unroll
            for (int i = 0; i < 8; ++i) { b[i] = blo[j][i]; b[i + 8] = bhi[j][i]; }
            acc[j] = __builtin_amdgcn_wmma_f32_16x16x32_bf16(
                false, a, false, b, (short)0, acc[j], false, false);
        }
    }

    const int row0 = tileM * 16 + half * 8;   // D layout: VGPR r holds M = r + 8*half
#pragma unroll
    for (int j = 0; j < 4; ++j) {
        const int tN = tileN0 + j;
        if (tN >= nTiles) break;
        const int col = tN * 16 + rc;
        const float bv = bias ? bias[col] : 0.0f;
#pragma unroll
        for (int r = 0; r < 8; ++r) {
            float v = acc[j][r] + bv;
            size_t idx = (size_t)(row0 + r) * ldD + col;
            if (OUT_BF16) Db[idx] = (__bf16)v; else Df[idx] = v;
        }
    }
}

// -------------------------------------------------------- attention passes -
__device__ __forceinline__ int f32_ord(float f) {
    int b = __float_as_int(f);
    return b >= 0 ? b : (b ^ 0x7fffffff);
}
__device__ __forceinline__ float ord_f32(int b) {
    return __int_as_float(b >= 0 ? b : (b ^ 0x7fffffff));
}

// Pass A: logits[e,h] = <q[dst], k[src]+e> / sqrt(C); atomic segment max.
__global__ void k_attn_logits(const float* __restrict__ qkv, int ld, int qoff, int koff,
                              const __bf16* __restrict__ ebf,
                              const int* __restrict__ src, const int* __restrict__ dst,
                              float* __restrict__ logits, int* __restrict__ segmax, int E_) {
    int t = blockIdx.x * blockDim.x + threadIdx.x;
    if (t >= E_ * 8) return;
    int e = t >> 3, h = t & 7;
    int sn = src[e], dn = dst[e];
    const float*  qp = qkv + (size_t)dn * ld + qoff + h * 32;
    const float*  kp = qkv + (size_t)sn * ld + koff + h * 32;
    const __bf16* ep = ebf + (size_t)e * 256 + h * 32;
    float s = 0.0f;
#pragma unroll
    for (int c = 0; c < 32; ++c) s += qp[c] * (kp[c] + (float)ep[c]);
    float l = s * 0.17677669529663687f;   // 1/sqrt(32)
    logits[t] = l;
    atomicMax(&segmax[dn * 8 + h], f32_ord(l));
}

// Pass B: ex = exp(l - max); accumulate denom; store ex in place of logits.
__global__ void k_attn_exp(const int* __restrict__ dst,
                           float* __restrict__ logits,
                           const int* __restrict__ segmax,
                           float* __restrict__ denom, int E_) {
    int t = blockIdx.x * blockDim.x + threadIdx.x;
    if (t >= E_ * 8) return;
    int e = t >> 3, h = t & 7;
    int dn = dst[e];
    float ex = __expf(logits[t] - ord_f32(segmax[dn * 8 + h]));
    logits[t] = ex;
    atomicAdd(&denom[dn * 8 + h], ex);
}

// Pass C: attn_out[dst] += alpha * (v[src] + e).
__global__ void k_attn_scatter(const float* __restrict__ qkv, int ld, int voff,
                               const __bf16* __restrict__ ebf,
                               const int* __restrict__ src, const int* __restrict__ dst,
                               const float* __restrict__ ex, const float* __restrict__ denom,
                               float* __restrict__ attn, int E_) {
    int t = blockIdx.x * blockDim.x + threadIdx.x;
    if (t >= E_ * 8) return;
    int e = t >> 3, h = t & 7;
    int sn = src[e], dn = dst[e];
    float alpha = ex[t] / (denom[dn * 8 + h] + 1e-16f);
    const float*  vp = qkv + (size_t)sn * ld + voff + h * 32;
    const __bf16* ep = ebf + (size_t)e * 256 + h * 32;
    float* op = attn + (size_t)dn * 256 + h * 32;
#pragma unroll
    for (int c = 0; c < 32; ++c)
        atomicAdd(&op[c], alpha * (vp[c] + (float)ep[c]));
}

// ---------------------------------------------------------- epilogues ------
// Layer 1: h = LayerNorm(ReLU(attn + skip)) -> bf16. One wave per node.
__global__ void k_epilogue1(const float* __restrict__ attn, const float* __restrict__ qkvs,
                            const float* __restrict__ g, const float* __restrict__ b,
                            __bf16* __restrict__ hbf, int N_) {
    int node = blockIdx.x * 8 + (threadIdx.x >> 5);
    if (node >= N_) return;
    int lane = threadIdx.x & 31;
    int base = lane * 8;
    const float* ap = attn + (size_t)node * 256;
    const float* sp = qkvs + (size_t)node * 1024 + 768;   // skip1 segment
    float v[8], s = 0.0f, s2 = 0.0f;
#pragma unroll
    for (int i = 0; i < 8; ++i) {
        float x = ap[base + i] + sp[base + i];
        x = fmaxf(x, 0.0f);
        v[i] = x; s += x; s2 += x * x;
    }
#pragma unroll
    for (int m = 16; m >= 1; m >>= 1) {
        s  += __shfl_xor(s,  m, 32);
        s2 += __shfl_xor(s2, m, 32);
    }
    float mu  = s * (1.0f / 256.0f);
    float var = s2 * (1.0f / 256.0f) - mu * mu;
    float inv = rsqrtf(var + 1e-5f);
#pragma unroll
    for (int i = 0; i < 8; ++i) {
        int c = base + i;
        hbf[(size_t)node * 256 + c] = (__bf16)((v[i] - mu) * inv * g[c] + b[c]);
    }
}

// Layer 2: y = ReLU(mean_h(attn) + skip2) @ wc + bc. One wave per node, lane=c.
__global__ void k_epilogue2(const float* __restrict__ attn, const float* __restrict__ qkv2,
                            const float* __restrict__ wc, const float* __restrict__ bc,
                            float* __restrict__ out, int N_) {
    int node = blockIdx.x * 8 + (threadIdx.x >> 5);
    if (node >= N_) return;
    int c = threadIdx.x & 31;
    const float* ap = attn + (size_t)node * 256;
    float v = 0.0f;
#pragma unroll
    for (int h = 0; h < 8; ++h) v += ap[h * 32 + c];
    v = v * 0.125f + qkv2[(size_t)node * 800 + 768 + c];  // mean over heads + skip2(+bias)
    v = fmaxf(v, 0.0f);
    float r0 = v * wc[c * 2 + 0];
    float r1 = v * wc[c * 2 + 1];
#pragma unroll
    for (int m = 16; m >= 1; m >>= 1) {
        r0 += __shfl_xor(r0, m, 32);
        r1 += __shfl_xor(r1, m, 32);
    }
    if (c == 0) {
        out[(size_t)node * 2 + 0] = r0 + bc[0];
        out[(size_t)node * 2 + 1] = r1 + bc[1];
    }
}

// ------------------------------------------------------------- launcher ----
extern "C" void kernel_launch(void* const* d_in, const int* in_sizes, int n_in,
                              void* d_out, int out_size, void* d_ws, size_t ws_size,
                              hipStream_t stream) {
    const float* x      = (const float*)d_in[0];
    const int*   eidx   = (const int*)  d_in[1];
    const float* eattr  = (const float*)d_in[2];
    const float* wq1    = (const float*)d_in[3];
    const float* bq1    = (const float*)d_in[4];
    const float* wk1    = (const float*)d_in[5];
    const float* bk1    = (const float*)d_in[6];
    const float* wv1    = (const float*)d_in[7];
    const float* bv1    = (const float*)d_in[8];
    const float* we1    = (const float*)d_in[9];
    const float* wskip1 = (const float*)d_in[10];
    const float* bskip1 = (const float*)d_in[11];
    const float* g1     = (const float*)d_in[12];
    const float* b1     = (const float*)d_in[13];
    const float* wq2    = (const float*)d_in[14];
    const float* bq2    = (const float*)d_in[15];
    const float* wk2    = (const float*)d_in[16];
    const float* bk2    = (const float*)d_in[17];
    const float* wv2    = (const float*)d_in[18];
    const float* bv2    = (const float*)d_in[19];
    const float* we2    = (const float*)d_in[20];
    const float* wskip2 = (const float*)d_in[21];
    const float* bskip2 = (const float*)d_in[22];
    const float* wc     = (const float*)d_in[23];
    const float* bc     = (const float*)d_in[24];

    const int N  = in_sizes[0] / 128;
    const int E  = in_sizes[2] / 32;
    const int* src = eidx;
    const int* dst = eidx + E;

    // ---- workspace carving ----
    char* p = (char*)d_ws;
    auto alloc = [&](size_t bytes) -> void* {
        void* r = (void*)p;
        p += (bytes + 255) & ~(size_t)255;
        return r;
    };
    __bf16* xbf   = (__bf16*)alloc((size_t)N * 128 * 2);
    __bf16* eabf  = (__bf16*)alloc((size_t)E * 32 * 2);
    __bf16* W1t   = (__bf16*)alloc((size_t)1024 * 128 * 2);
    __bf16* Wet   = (__bf16*)alloc((size_t)512 * 32 * 2);
    __bf16* W2t   = (__bf16*)alloc((size_t)800 * 256 * 2);
    float*  bias1 = (float*) alloc(1024 * 4);
    float*  bias2 = (float*) alloc(800 * 4);
    float*  QKVS  = (float*) alloc((size_t)N * 1024 * 4);   // reused as N x 800 in layer 2
    __bf16* Ebf   = (__bf16*)alloc((size_t)E * 256 * 2);    // e1 then e2
    float*  logit = (float*) alloc((size_t)E * 8 * 4);      // logits -> ex in place
    int*    smax  = (int*)   alloc((size_t)N * 8 * 4);
    float*  denom = (float*) alloc((size_t)N * 8 * 4);
    float*  attn  = (float*) alloc((size_t)N * 256 * 4);
    __bf16* hbf   = (__bf16*)alloc((size_t)N * 256 * 2);
    (void)ws_size; (void)n_in;

    const int TB = 256;
    auto blocks = [&](long long n) { return (int)((n + TB - 1) / TB); };

    // ---- packing (cheap, per call) ----
    k_f32_to_bf16<<<blocks((long long)N * 128), TB, 0, stream>>>(x, xbf, N * 128);
    k_f32_to_bf16<<<blocks((long long)E * 32),  TB, 0, stream>>>(eattr, eabf, E * 32);

    k_pack_wt<<<blocks(128 * 256), TB, 0, stream>>>(wq1,    W1t + (size_t)0   * 128, 128, 256);
    k_pack_wt<<<blocks(128 * 256), TB, 0, stream>>>(wk1,    W1t + (size_t)256 * 128, 128, 256);
    k_pack_wt<<<blocks(128 * 256), TB, 0, stream>>>(wv1,    W1t + (size_t)512 * 128, 128, 256);
    k_pack_wt<<<blocks(128 * 256), TB, 0, stream>>>(wskip1, W1t + (size_t)768 * 128, 128, 256);
    k_copy_f32<<<1, 256, 0, stream>>>(bq1,    bias1 + 0,   256);
    k_copy_f32<<<1, 256, 0, stream>>>(bk1,    bias1 + 256, 256);
    k_copy_f32<<<1, 256, 0, stream>>>(bv1,    bias1 + 512, 256);
    k_copy_f32<<<1, 256, 0, stream>>>(bskip1, bias1 + 768, 256);

    k_pack_wt<<<blocks(32 * 256), TB, 0, stream>>>(we1, Wet + (size_t)0   * 32, 32, 256);
    k_pack_wt<<<blocks(32 * 256), TB, 0, stream>>>(we2, Wet + (size_t)256 * 32, 32, 256);

    k_pack_wt<<<blocks(256 * 256), TB, 0, stream>>>(wq2,    W2t + (size_t)0   * 256, 256, 256);
    k_pack_wt<<<blocks(256 * 256), TB, 0, stream>>>(wk2,    W2t + (size_t)256 * 256, 256, 256);
    k_pack_wt<<<blocks(256 * 256), TB, 0, stream>>>(wv2,    W2t + (size_t)512 * 256, 256, 256);
    k_pack_wt<<<blocks(256 * 32),  TB, 0, stream>>>(wskip2, W2t + (size_t)768 * 256, 256, 32);
    k_copy_f32<<<1, 256, 0, stream>>>(bq2,    bias2 + 0,   256);
    k_copy_f32<<<1, 256, 0, stream>>>(bk2,    bias2 + 256, 256);
    k_copy_f32<<<1, 256, 0, stream>>>(bv2,    bias2 + 512, 256);
    k_copy_f32<<<1, 32, 0, stream>>>(bskip2,  bias2 + 768, 32);

    const int Mt = (N + 15) / 16;       // 3125
    const int Et = (E + 15) / 16;       // 25000

    // ================= layer 1 =================
    // QKVS[N x 1024] = xbf @ [wq1|wk1|wv1|wskip1] + biases
    k_wmma_gemm<false><<<dim3(Mt, 1024 / 256), 128, 0, stream>>>(
        xbf, W1t, bias1, QKVS, nullptr, N, 128, 1024, 1024);
    // Ebf[E x 256] = eattr @ we1  (bf16 out, no bias)
    k_wmma_gemm<true><<<dim3(Et, 1), 128, 0, stream>>>(
        eabf, Wet, nullptr, nullptr, Ebf, E, 32, 256, 256);

    k_fill_u32<<<blocks((long long)N * 8), TB, 0, stream>>>((unsigned*)smax, 0x80000000u, N * 8);
    k_fill_u32<<<blocks((long long)N * 8), TB, 0, stream>>>((unsigned*)denom, 0u, N * 8);
    k_fill_u32<<<blocks((long long)N * 256), TB, 0, stream>>>((unsigned*)attn, 0u, N * 256);

    k_attn_logits<<<blocks((long long)E * 8), TB, 0, stream>>>(
        QKVS, 1024, 0, 256, Ebf, src, dst, logit, smax, E);
    k_attn_exp<<<blocks((long long)E * 8), TB, 0, stream>>>(dst, logit, smax, denom, E);
    k_attn_scatter<<<blocks((long long)E * 8), TB, 0, stream>>>(
        QKVS, 1024, 512, Ebf, src, dst, logit, denom, attn, E);

    k_epilogue1<<<(N + 7) / 8, 256, 0, stream>>>(attn, QKVS, g1, b1, hbf, N);

    // ================= layer 2 =================
    // QKVS[N x 800] = hbf @ [wq2|wk2|wv2|wskip2] + biases
    k_wmma_gemm<false><<<dim3(Mt, (800 + 255) / 256), 128, 0, stream>>>(
        hbf, W2t, bias2, QKVS, nullptr, N, 256, 800, 800);
    // Ebf[E x 256] = eattr @ we2
    k_wmma_gemm<true><<<dim3(Et, 1), 128, 0, stream>>>(
        eabf, Wet + (size_t)256 * 32, nullptr, nullptr, Ebf, E, 32, 256, 256);

    k_fill_u32<<<blocks((long long)N * 8), TB, 0, stream>>>((unsigned*)smax, 0x80000000u, N * 8);
    k_fill_u32<<<blocks((long long)N * 8), TB, 0, stream>>>((unsigned*)denom, 0u, N * 8);
    k_fill_u32<<<blocks((long long)N * 256), TB, 0, stream>>>((unsigned*)attn, 0u, N * 256);

    k_attn_logits<<<blocks((long long)E * 8), TB, 0, stream>>>(
        QKVS, 800, 0, 256, Ebf, src, dst, logit, smax, E);
    k_attn_exp<<<blocks((long long)E * 8), TB, 0, stream>>>(dst, logit, smax, denom, E);
    k_attn_scatter<<<blocks((long long)E * 8), TB, 0, stream>>>(
        QKVS, 800, 512, Ebf, src, dst, logit, denom, attn, E);

    k_epilogue2<<<(N + 7) / 8, 256, 0, stream>>>(attn, QKVS, wc, bc, (float*)d_out, N);
    (void)out_size;
}
